// OdorGATv5_10771777978539
// MI455X (gfx1250) — matrix operand order repaired and checked by reading, not compiled, not run on hardware
//
#include <hip/hip_runtime.h>
#include <hip/hip_bf16.h>

// ---------------------------------------------------------------------------
// OdorGAT pipeline for MI455X (gfx1250, wave32).
// Dense GEMMs -> v_wmma_f32_16x16x32_f16, f16 weights pre-converted in ws,
// LDS A-panel stored in fragment order (one ds_load_b128 pair per fragment).
// Edge softmax/aggregation -> wave-per-edge coalesced gather + f32 atomics.
// ---------------------------------------------------------------------------

#define N_NODES 100000
#define N_EDGES 300000
#define N_GRAPHS 4096
#define BERT_DIM 384

typedef __attribute__((ext_vector_type(16))) _Float16 v16h;
typedef __attribute__((ext_vector_type(8)))  float    v8f;

static inline long cdiv(long a, long b) { return (a + b - 1) / b; }

// ---- orderable float <-> uint key (for atomicMax on floats) ----------------
__device__ __forceinline__ unsigned fkey(float f) {
  unsigned u = __float_as_uint(f);
  return (u & 0x80000000u) ? ~u : (u | 0x80000000u);
}
__device__ __forceinline__ float fdecode(unsigned k) {
  unsigned u = (k & 0x80000000u) ? (k & 0x7FFFFFFFu) : ~k;
  return __uint_as_float(u);
}

// ---- f32 -> f16 weight pre-conversion --------------------------------------
__global__ void k_f2h(const float* __restrict__ src, _Float16* __restrict__ dst, long n) {
  long i = (long)blockIdx.x * blockDim.x + threadIdx.x;
  if (i < n) dst[i] = (_Float16)src[i];
}

// ---------------------------------------------------------------------------
// WMMA GEMM: C[M,N] = A[M,K](f32) * B16[K,N](f16), f32 accum.
// Requirements (all call sites satisfy): M % 16 == 0, Nout % 64 == 0 or ==64,
// K in {256,384} (template, fully unrolled k-loop).
// Block = 128 threads (4 waves); wave computes a 16 x 64 strip (4 WMMA tiles).
// A panel staged in LDS in *fragment order*: halfword
//   sA[(kblk*32 + fraglane)*16 + slot], so each lane's A fragment is one
//   contiguous 32-byte LDS read.  Fragment layout per ISA 16-bit A matrix:
//   lane L holds row (L&15), K = kg+{0..7} (slots 0-7) and kg+16+{0..7}
//   (slots 8-15), kg = (L>>4)*8.
// ---------------------------------------------------------------------------
#define GEMM_TB 128

template <int K>
__global__ __launch_bounds__(GEMM_TB)
void k_gemm_wmma(const float* __restrict__ A, const _Float16* __restrict__ B16,
                 float* __restrict__ C, int M, int Nout) {
  __shared__ __attribute__((aligned(32))) _Float16 sA[16 * K];
  const int m0 = blockIdx.x * 16;
  const int tid = threadIdx.x;
  const int lane = tid & 31;
  const int wave = tid >> 5;

  // Stage + swizzle A tile (16 x K) into fragment-ordered f16 LDS.
  for (int i = tid; i < 16 * K; i += GEMM_TB) {
    int r = i / K, k = i - r * K;              // K is compile-time: fast div
    int kb = k >> 5, kk = k & 31;
    int lt = r + (((kk >> 3) & 1) << 4);       // fragment lane
    int slot = (kk & 7) | (((kk >> 4) & 1) << 3);
    sA[(kb * 32 + lt) * 16 + slot] = (_Float16)A[(size_t)(m0 + r) * K + k];
  }
  __syncthreads();

  const int n0 = wave * 64;
  if (n0 >= Nout) return;                      // wave-uniform: EXEC all-ones

  v8f acc0 = {}, acc1 = {}, acc2 = {}, acc3 = {};
#pragma unroll
  for (int kb = 0; kb < K / 32; ++kb) {
    v16h a = *(const v16h*)&sA[(kb * 32 + lane) * 16];
    const _Float16* bp = B16 + (size_t)(kb * 32 + (lane >> 4) * 8) * Nout + n0 + (lane & 15);
    v16h b0, b1, b2, b3;
#pragma unroll
    for (int s = 0; s < 16; ++s) {
      const _Float16* p = bp + (size_t)((s & 7) + ((s >> 3) << 4)) * Nout;
      b0[s] = p[0];
      b1[s] = p[16];
      b2[s] = p[32];
      b3[s] = p[48];
    }
    acc0 = __builtin_amdgcn_wmma_f32_16x16x32_f16(false, a, false, b0, (short)0, acc0, false, false);
    acc1 = __builtin_amdgcn_wmma_f32_16x16x32_f16(false, a, false, b1, (short)0, acc1, false, false);
    acc2 = __builtin_amdgcn_wmma_f32_16x16x32_f16(false, a, false, b2, (short)0, acc2, false, false);
    acc3 = __builtin_amdgcn_wmma_f32_16x16x32_f16(false, a, false, b3, (short)0, acc3, false, false);
  }

  // C/D layout: VGPR r, lanes 0-15 -> M=r,N=lane ; lanes 16-31 -> M=8+r.
  const int col = lane & 15;
  const int rb  = (lane >> 4) * 8;
  float* cp = C + (size_t)(m0 + rb) * Nout + n0 + col;
#pragma unroll
  for (int r = 0; r < 8; ++r) {
    float* row = cp + (size_t)r * Nout;
    row[0]  = acc0[r];
    row[16] = acc1[r];
    row[32] = acc2[r];
    row[48] = acc3[r];
  }
}

// ---- tiny-K GEMM (layer1 K=9) ----------------------------------------------
__global__ void k_gemm_small(const float* __restrict__ A, const float* __restrict__ B,
                             float* __restrict__ C, int M, int K, int Nout) {
  long idx = (long)blockIdx.x * blockDim.x + threadIdx.x;
  if (idx >= (long)M * Nout) return;
  int m = (int)(idx / Nout), j = (int)(idx - (long)m * Nout);
  float s = 0.0f;
  for (int k = 0; k < K; ++k) s += A[(size_t)m * K + k] * B[(size_t)k * Nout + j];
  C[idx] = s;
}

__global__ void k_head_sigmoid(const float* __restrict__ A, const float* __restrict__ Wt,
                               const float* __restrict__ bias, float* __restrict__ out,
                               int M, int K, int Nout) {
  long idx = (long)blockIdx.x * blockDim.x + threadIdx.x;
  if (idx >= (long)M * Nout) return;
  int m = (int)(idx / Nout), j = (int)(idx - (long)m * Nout);
  float s = bias[j];
  for (int k = 0; k < K; ++k) s += A[(size_t)m * K + k] * Wt[(size_t)k * Nout + j];
  out[idx] = 1.0f / (1.0f + __expf(-s));
}

// ---- self-loop attr: counts + per-dst edge_attr sums -----------------------
__global__ void k_edge_accum(const int* __restrict__ ei, const float* __restrict__ eattr,
                             float* __restrict__ cnt, float* __restrict__ lsum) {
  int e = blockIdx.x * blockDim.x + threadIdx.x;
  if (e >= N_EDGES) return;
  int d = ei[N_EDGES + e];
  atomicAdd(&cnt[d], 1.0f);
  for (int j = 0; j < 3; ++j) atomicAdd(&lsum[(size_t)d * 3 + j], eattr[(size_t)e * 3 + j]);
}
__global__ void k_loop_div(float* __restrict__ lat, const float* __restrict__ cnt) {
  int i = blockIdx.x * blockDim.x + threadIdx.x;
  if (i >= N_NODES * 3) return;
  lat[i] /= fmaxf(cnt[i / 3], 1.0f);
}

// ---- edge-attention coefficient matrix: Me[j,h] = We[j,h,:].att_edge[h,:] --
__global__ void k_me(const float* __restrict__ We, const float* __restrict__ att_e,
                     float* __restrict__ Me, int H, int C) {
  int t = threadIdx.x;
  if (t >= 3 * H) return;
  int j = t / H, h = t % H, HC = H * C;
  float s = 0.0f;
  for (int c = 0; c < C; ++c) s += We[(size_t)j * HC + h * C + c] * att_e[h * C + c];
  Me[t] = s;
}

// ---- per-node attention scalars: a_src, a_dst ------------------------------
__global__ void k_nodecoef(const float* __restrict__ h, const float* __restrict__ att_s,
                           const float* __restrict__ att_d, float* __restrict__ a_src,
                           float* __restrict__ a_dst, int H, int C) {
  long idx = (long)blockIdx.x * blockDim.x + threadIdx.x;
  if (idx >= (long)N_NODES * H) return;
  int n = (int)(idx / H), hh = (int)(idx % H);
  const float* hp = h + (size_t)n * H * C + hh * C;
  float s1 = 0.0f, s2 = 0.0f;
  for (int c = 0; c < C; ++c) { s1 += hp[c] * att_s[hh * C + c]; s2 += hp[c] * att_d[hh * C + c]; }
  a_src[idx] = s1; a_dst[idx] = s2;
}

// ---- alpha = leaky_relu(a_src[s]+a_dst[d]+a_e); segment max by dst ---------
__global__ void k_alpha_max(const int* __restrict__ ei, const float* __restrict__ eattr,
                            const float* __restrict__ lat, const float* __restrict__ Me,
                            const float* __restrict__ a_src, const float* __restrict__ a_dst,
                            float* __restrict__ alpha, unsigned* __restrict__ amaxk, int H) {
  const int Etot = N_EDGES + N_NODES;
  long idx = (long)blockIdx.x * blockDim.x + threadIdx.x;
  if (idx >= (long)Etot * H) return;
  int e = (int)(idx / H), hh = (int)(idx % H);
  int s, d; const float* ea;
  if (e < N_EDGES) { s = ei[e]; d = ei[N_EDGES + e]; ea = eattr + (size_t)e * 3; }
  else             { s = d = e - N_EDGES;            ea = lat   + (size_t)(e - N_EDGES) * 3; }
  float ae = ea[0] * Me[0 * H + hh] + ea[1] * Me[1 * H + hh] + ea[2] * Me[2 * H + hh];
  float al = a_src[(size_t)s * H + hh] + a_dst[(size_t)d * H + hh] + ae;
  al = (al > 0.0f) ? al : 0.2f * al;
  alpha[idx] = al;
  atomicMax(&amaxk[(size_t)d * H + hh], fkey(al));
}

// ---- w = exp(alpha - max); segment sum by dst ------------------------------
__global__ void k_expsum(const int* __restrict__ ei, float* __restrict__ alpha,
                         const unsigned* __restrict__ amaxk, float* __restrict__ asum, int H) {
  const int Etot = N_EDGES + N_NODES;
  long idx = (long)blockIdx.x * blockDim.x + threadIdx.x;
  if (idx >= (long)Etot * H) return;
  int e = (int)(idx / H), hh = (int)(idx % H);
  int d = (e < N_EDGES) ? ei[N_EDGES + e] : (e - N_EDGES);
  float w = __expf(alpha[idx] - fdecode(amaxk[(size_t)d * H + hh]));
  alpha[idx] = w;
  atomicAdd(&asum[(size_t)d * H + hh], w);
}

// ---- aggregation: one wave per edge; out[dst] += w * h[src] ----------------
__global__ void k_aggregate(const float* __restrict__ h, const int* __restrict__ ei,
                            const float* __restrict__ alpha, const float* __restrict__ asum,
                            float* __restrict__ out, int H, int C) {
  const int Etot = N_EDGES + N_NODES;
  long gw = ((long)blockIdx.x * blockDim.x + threadIdx.x) >> 5;
  int lane = threadIdx.x & 31;
  if (gw >= Etot) return;
  int s, d;
  if (gw < N_EDGES) { s = ei[gw]; d = ei[N_EDGES + gw]; }
  else              { s = d = (int)(gw - N_EDGES); }
  int HC = H * C, per = HC >> 5;
  int c0 = lane * per, head = c0 / C;
  float w = alpha[gw * H + head] / (asum[(size_t)d * H + head] + 1e-16f);
  const float* hp = h + (size_t)s * HC + c0;
  float* op = out + (size_t)d * HC + c0;
  for (int i = 0; i < per; ++i) atomicAdd(&op[i], w * hp[i]);
}

// ---- fused bias + (mode0: LN->ELU) | (mode1: GELU->LN); wave per row -------
__global__ void k_act_ln(const float* __restrict__ in, int in_stride,
                         const float* __restrict__ bias, const float* __restrict__ gam,
                         const float* __restrict__ bet, float* __restrict__ out,
                         int out_stride, int rows, int cols, int mode) {
  long gw = ((long)blockIdx.x * blockDim.x + threadIdx.x) >> 5;
  int lane = threadIdx.x & 31;
  if (gw >= rows) return;
  int per = cols >> 5;                       // <= 8
  float v[8];
  const float* ip = in + (size_t)gw * in_stride + lane * per;
  float s = 0.0f;
  for (int i = 0; i < per; ++i) {
    float x = ip[i] + (bias ? bias[lane * per + i] : 0.0f);
    if (mode == 1) x = 0.5f * x * (1.0f + erff(x * 0.7071067811865475f));  // exact GELU
    v[i] = x; s += x;
  }
  for (int m = 16; m; m >>= 1) s += __shfl_xor(s, m, 32);
  float mean = s / cols, q = 0.0f;
  for (int i = 0; i < per; ++i) { float dd = v[i] - mean; q += dd * dd; }
  for (int m = 16; m; m >>= 1) q += __shfl_xor(q, m, 32);
  float inv = rsqrtf(q / cols + 1e-5f);
  float* op = out + (size_t)gw * out_stride + lane * per;
  for (int i = 0; i < per; ++i) {
    int c = lane * per + i;
    float y = (v[i] - mean) * inv * gam[c] + bet[c];
    if (mode == 0) y = (y > 0.0f) ? y : (__expf(y) - 1.0f);                // ELU
    op[i] = y;
  }
}

// ---- graph pooling (mean + max over batch segments) ------------------------
__global__ void k_pool(const float* __restrict__ h, const int* __restrict__ batch,
                       float* __restrict__ psum, float* __restrict__ pcnt,
                       unsigned* __restrict__ pmaxk) {
  long gw = ((long)blockIdx.x * blockDim.x + threadIdx.x) >> 5;
  int lane = threadIdx.x & 31;
  if (gw >= N_NODES) return;
  int g = batch[gw], c = lane * 2;
  for (int i = 0; i < 2; ++i) {
    float v = h[gw * 64 + c + i];
    atomicAdd(&psum[(size_t)g * 64 + c + i], v);
    atomicMax(&pmaxk[(size_t)g * 64 + c + i], fkey(v));
  }
  if (lane == 0) atomicAdd(&pcnt[g], 1.0f);
}
__global__ void k_pool_fin(const float* __restrict__ psum, const float* __restrict__ pcnt,
                           const unsigned* __restrict__ pmaxk, float* __restrict__ z) {
  int idx = blockIdx.x * blockDim.x + threadIdx.x;
  if (idx >= N_GRAPHS * 64) return;
  int g = idx / 64, c = idx % 64;
  float cn = pcnt[g];
  z[(size_t)g * 384 + c]      = psum[idx] / fmaxf(cn, 1.0f);
  z[(size_t)g * 384 + 64 + c] = (cn > 0.0f) ? fdecode(pmaxk[idx]) : 0.0f;
}

// ---------------------------------------------------------------------------
// Host-side GAT attention stage (after h = X @ W is in hbuf).
// ---------------------------------------------------------------------------
static void gat_attention(hipStream_t st, const int* ei, const float* eattr,
                          const float* lat, const float* We, const float* att_e,
                          const float* att_s, const float* att_d,
                          const float* hbuf, float* outbuf,
                          float* a_src, float* a_dst, float* alpha,
                          unsigned* amaxk, float* asum, float* Me,
                          const float* bias, const float* lng, const float* lnb,
                          int H, int C) {
  const long Etot = N_EDGES + N_NODES;
  k_me<<<1, 32, 0, st>>>(We, att_e, Me, H, C);
  k_nodecoef<<<(int)cdiv((long)N_NODES * H, 256), 256, 0, st>>>(hbuf, att_s, att_d, a_src, a_dst, H, C);
  hipMemsetAsync(amaxk, 0, (size_t)N_NODES * H * 4, st);
  hipMemsetAsync(asum,  0, (size_t)N_NODES * H * 4, st);
  k_alpha_max<<<(int)cdiv(Etot * H, 256), 256, 0, st>>>(ei, eattr, lat, Me, a_src, a_dst, alpha, amaxk, H);
  k_expsum<<<(int)cdiv(Etot * H, 256), 256, 0, st>>>(ei, alpha, amaxk, asum, H);
  hipMemsetAsync(outbuf, 0, (size_t)N_NODES * H * C * 4, st);
  k_aggregate<<<(int)cdiv(Etot * 32, 256), 256, 0, st>>>(hbuf, ei, alpha, asum, outbuf, H, C);
  k_act_ln<<<(int)cdiv((long)N_NODES * 32, 256), 256, 0, st>>>(outbuf, H * C, bias, lng, lnb,
                                                               outbuf, H * C, N_NODES, H * C, 0);
}

// ---------------------------------------------------------------------------
extern "C" void kernel_launch(void* const* d_in, const int* in_sizes, int n_in,
                              void* d_out, int out_size, void* d_ws, size_t ws_size,
                              hipStream_t stream) {
  (void)in_sizes; (void)n_in; (void)out_size; (void)ws_size;
  const float* x     = (const float*)d_in[0];
  const int*   ei    = (const int*)d_in[1];
  const float* eattr = (const float*)d_in[2];
  const int*   batch = (const int*)d_in[3];
  const float* bert  = (const float*)d_in[4];
  // gat1..gat3: W, We, att_src, att_dst, att_edge, bias
  const float* g1W = (const float*)d_in[5],  *g1We = (const float*)d_in[6],
             *g1as = (const float*)d_in[7],  *g1ad = (const float*)d_in[8],
             *g1ae = (const float*)d_in[9],  *g1b  = (const float*)d_in[10];
  const float* g2W = (const float*)d_in[11], *g2We = (const float*)d_in[12],
             *g2as = (const float*)d_in[13], *g2ad = (const float*)d_in[14],
             *g2ae = (const float*)d_in[15], *g2b  = (const float*)d_in[16];
  const float* g3W = (const float*)d_in[17], *g3We = (const float*)d_in[18],
             *g3as = (const float*)d_in[19], *g3ad = (const float*)d_in[20],
             *g3ae = (const float*)d_in[21], *g3b  = (const float*)d_in[22];
  const float* ln1g = (const float*)d_in[23], *ln1b = (const float*)d_in[24];
  const float* ln2g = (const float*)d_in[25], *ln2b = (const float*)d_in[26];
  const float* ln3g = (const float*)d_in[27], *ln3b = (const float*)d_in[28];
  const float* bpW = (const float*)d_in[29], *bpb = (const float*)d_in[30],
             *bpg  = (const float*)d_in[31], *bpbe = (const float*)d_in[32];
  const float* f1W = (const float*)d_in[33], *f1b = (const float*)d_in[34],
             *f1g  = (const float*)d_in[35], *f1be = (const float*)d_in[36];
  const float* f2W = (const float*)d_in[37], *f2b = (const float*)d_in[38],
             *f2g  = (const float*)d_in[39], *f2be = (const float*)d_in[40];
  const float* f3W = (const float*)d_in[41], *f3b = (const float*)d_in[42];

  // workspace layout (float granularity, 256B aligned chunks)
  float* W = (float*)d_ws;
  size_t o = 0;
  auto alloc = [&](size_t nf) { size_t r = o; o += (nf + 63) & ~(size_t)63; return r; };
  float*    bufA  = W + alloc((size_t)N_NODES * 256);   // node features / aggregation out
  float*    bufB  = W + alloc((size_t)N_NODES * 256);   // GEMM result h (messages)
  float*    lat   = W + alloc((size_t)N_NODES * 3);     // self-loop edge attr
  float*    cnt   = W + alloc(N_NODES);
  float*    a_src = W + alloc((size_t)N_NODES * 4);
  float*    a_dst = W + alloc((size_t)N_NODES * 4);
  float*    alpha = W + alloc((size_t)(N_EDGES + N_NODES) * 4);
  unsigned* amaxk = (unsigned*)(W + alloc((size_t)N_NODES * 4));
  float*    asum  = W + alloc((size_t)N_NODES * 4);
  float*    Me    = W + alloc(64);
  float*    psum  = W + alloc((size_t)N_GRAPHS * 64);
  float*    pcnt  = W + alloc(N_GRAPHS);
  unsigned* pmaxk = (unsigned*)(W + alloc((size_t)N_GRAPHS * 64));
  float*    zbuf  = W + alloc((size_t)N_GRAPHS * 384);  // [mean64 | max64 | bert256]
  float*    t1    = W + alloc((size_t)N_GRAPHS * 256);
  float*    t1n   = W + alloc((size_t)N_GRAPHS * 256);
  float*    t2    = W + alloc((size_t)N_GRAPHS * 128);
  float*    t2n   = W + alloc((size_t)N_GRAPHS * 128);
  // f16 weight copies (halfword buffers carved from float space)
  _Float16* g2W16 = (_Float16*)(W + alloc(256 * 256 / 2));
  _Float16* g3W16 = (_Float16*)(W + alloc(256 * 64 / 2));
  _Float16* bpW16 = (_Float16*)(W + alloc(384 * 256 / 2));
  _Float16* f1W16 = (_Float16*)(W + alloc(384 * 256 / 2));
  _Float16* f2W16 = (_Float16*)(W + alloc(256 * 128 / 2));

  // ---- one-time weight conversion to f16 ----------------------------------
  k_f2h<<<(int)cdiv(256 * 256, 256), 256, 0, stream>>>(g2W, g2W16, 256 * 256);
  k_f2h<<<(int)cdiv(256 * 64, 256), 256, 0, stream>>>(g3W, g3W16, 256 * 64);
  k_f2h<<<(int)cdiv(384 * 256, 256), 256, 0, stream>>>(bpW, bpW16, 384 * 256);
  k_f2h<<<(int)cdiv(384 * 256, 256), 256, 0, stream>>>(f1W, f1W16, 384 * 256);
  k_f2h<<<(int)cdiv(256 * 128, 256), 256, 0, stream>>>(f2W, f2W16, 256 * 128);

  // ---- self-loop edge attributes (fill_value='mean'), computed once -------
  hipMemsetAsync(cnt, 0, (size_t)N_NODES * 4, stream);
  hipMemsetAsync(lat, 0, (size_t)N_NODES * 3 * 4, stream);
  k_edge_accum<<<(int)cdiv(N_EDGES, 256), 256, 0, stream>>>(ei, eattr, cnt, lat);
  k_loop_div<<<(int)cdiv((long)N_NODES * 3, 256), 256, 0, stream>>>(lat, cnt);

  // ---- layer 1: K=9 -> scalar GEMM, then attention -------------------------
  k_gemm_small<<<(int)cdiv((long)N_NODES * 256, 256), 256, 0, stream>>>(x, g1W, bufB, N_NODES, 9, 256);
  gat_attention(stream, ei, eattr, lat, g1We, g1ae, g1as, g1ad, bufB, bufA,
                a_src, a_dst, alpha, amaxk, asum, Me, g1b, ln1g, ln1b, 4, 64);

  // ---- layer 2: 100000x256x256 WMMA GEMM ----------------------------------
  k_gemm_wmma<256><<<(int)cdiv(N_NODES, 16), GEMM_TB, 0, stream>>>(bufA, g2W16, bufB, N_NODES, 256);
  gat_attention(stream, ei, eattr, lat, g2We, g2ae, g2as, g2ad, bufB, bufA,
                a_src, a_dst, alpha, amaxk, asum, Me, g2b, ln2g, ln2b, 4, 64);

  // ---- layer 3: 100000x256x64 WMMA GEMM (heads=1, mean == identity) -------
  k_gemm_wmma<256><<<(int)cdiv(N_NODES, 16), GEMM_TB, 0, stream>>>(bufA, g3W16, bufB, N_NODES, 64);
  gat_attention(stream, ei, eattr, lat, g3We, g3ae, g3as, g3ad, bufB, bufA,
                a_src, a_dst, alpha, amaxk, asum, Me, g3b, ln3g, ln3b, 1, 64);

  // ---- graph pooling: mean + max into zbuf[:, 0:128] ----------------------
  hipMemsetAsync(psum,  0, (size_t)N_GRAPHS * 64 * 4, stream);
  hipMemsetAsync(pcnt,  0, (size_t)N_GRAPHS * 4, stream);
  hipMemsetAsync(pmaxk, 0, (size_t)N_GRAPHS * 64 * 4, stream);
  k_pool<<<(int)cdiv((long)N_NODES * 32, 256), 256, 0, stream>>>(bufA, batch, psum, pcnt, pmaxk);
  k_pool_fin<<<(int)cdiv((long)N_GRAPHS * 64, 256), 256, 0, stream>>>(psum, pcnt, pmaxk, zbuf);

  // ---- BERT projection: 4096x384x256 WMMA, GELU+LN into zbuf[:, 128:384] --
  k_gemm_wmma<384><<<(int)cdiv(N_GRAPHS, 16), GEMM_TB, 0, stream>>>(bert, bpW16, t1, N_GRAPHS, 256);
  k_act_ln<<<(int)cdiv((long)N_GRAPHS * 32, 256), 256, 0, stream>>>(t1, 256, bpb, bpg, bpbe,
                                                                    zbuf + 128, 384, N_GRAPHS, 256, 1);

  // ---- MLP head -----------------------------------------------------------
  k_gemm_wmma<384><<<(int)cdiv(N_GRAPHS, 16), GEMM_TB, 0, stream>>>(zbuf, f1W16, t1, N_GRAPHS, 256);
  k_act_ln<<<(int)cdiv((long)N_GRAPHS * 32, 256), 256, 0, stream>>>(t1, 256, f1b, f1g, f1be,
                                                                    t1n, 256, N_GRAPHS, 256, 1);
  k_gemm_wmma<256><<<(int)cdiv(N_GRAPHS, 16), GEMM_TB, 0, stream>>>(t1n, f2W16, t2, N_GRAPHS, 128);
  k_act_ln<<<(int)cdiv((long)N_GRAPHS * 32, 256), 256, 0, stream>>>(t2, 128, f2b, f2g, f2be,
                                                                    t2n, 128, N_GRAPHS, 128, 1);
  k_head_sigmoid<<<(int)cdiv((long)N_GRAPHS * 22, 256), 256, 0, stream>>>(t2n, f3W, f3b,
                                                                          (float*)d_out,
                                                                          N_GRAPHS, 128, 22);
}
